// GNN_1932735283948
// MI455X (gfx1250) — compile-verified, matrix operand order checked by reading, not run on hardware
//
#include <hip/hip_runtime.h>

// ---------------- problem constants (match reference) ----------------
#define N_NODES   50000
#define N_EDGES   800000
#define D_IN      128
#define HID       256
#define N_CLASSES 10
#define N_GRAPHS  128

// ---------------- WMMA types ----------------
typedef __attribute__((ext_vector_type(16))) __bf16 v16bf;
typedef __attribute__((ext_vector_type(8)))  __bf16 v8bf;
typedef __attribute__((ext_vector_type(4)))  __bf16 v4bf;
typedef __attribute__((ext_vector_type(8)))  float  v8f;

// ---------------- GEMM tiling ----------------
#define BM 128
#define BN 64
#define BK 32
#define LDA (BK + 8)   // A tile LDS row stride (elements; 80B rows -> 16B-aligned frags)
#define LDBT (BK + 8)  // transposed-B tile LDS row stride (elements)

// =====================================================================
// zero-fill
// =====================================================================
__global__ void k_zero(float* __restrict__ p, long n) {
    long i = (long)blockIdx.x * blockDim.x + threadIdx.x;
    if (i < n) p[i] = 0.0f;
}

// =====================================================================
// edge scatter-add: agg[dst] += x[src]; one thread per (edge, float4)
// =====================================================================
template<int SH>
__global__ void k_scatter_add(const float* __restrict__ x,
                              const int* __restrict__ src,
                              const int* __restrict__ dst,
                              float* __restrict__ agg,
                              int n_edges, int D) {
    long tid = (long)blockIdx.x * blockDim.x + threadIdx.x;
    long total = (long)n_edges << SH;
    if (tid >= total) return;
    int e = (int)(tid >> SH);
    int c = (int)(tid & ((1 << SH) - 1));
    int s = src[e];
    int d = dst[e];
    const float4 v = *(const float4*)(x + (long)s * D + c * 4);
    float* p = agg + (long)d * D + c * 4;
    atomicAdd(p + 0, v.x);
    atomicAdd(p + 1, v.y);
    atomicAdd(p + 2, v.z);
    atomicAdd(p + 3, v.w);
}

// ---- fragment load: two 16B LDS reads, concat into v16bf ----
// element j <-> k = khalf + (j<8 ? j : j+8); lo run at khalf, hi run at khalf+16
static __device__ __forceinline__ v16bf load_frag(const __bf16* base /*row start*/, int khalf) {
    v8bf lo = *(const v8bf*)(base + khalf);
    v8bf hi = *(const v8bf*)(base + khalf + 16);
    return __builtin_shufflevector(lo, hi, 0, 1, 2, 3, 4, 5, 6, 7,
                                           8, 9, 10, 11, 12, 13, 14, 15);
}

static __device__ __forceinline__ v4bf cvt4(float4 f) {
    v4bf r;
    r[0] = (__bf16)f.x; r[1] = (__bf16)f.y; r[2] = (__bf16)f.z; r[3] = (__bf16)f.w;
    return r;
}

// =====================================================================
// Fused dual GEMM: C = act(A1@B1 + A2@B2 + bias)
//   A1,A2: [M,K] row-major f32   B1,B2: [K,N] row-major f32
//   bf16 WMMA (v_wmma_f32_16x16x32_bf16), f32 accumulate.
// Block: 256 threads (8 waves). Block tile 128x64; wave tile 32x32.
// B is staged TRANSPOSED in LDS so fragment reads are contiguous b128.
// =====================================================================
template<bool RELU>
__global__ __launch_bounds__(256)
void k_dual_gemm_bias(const float* __restrict__ A1, const float* __restrict__ A2,
                      const float* __restrict__ B1, const float* __restrict__ B2,
                      const float* __restrict__ bias, float* __restrict__ C,
                      int M, int K, int N) {
    __shared__ __bf16 sA1[BM * LDA];
    __shared__ __bf16 sA2[BM * LDA];
    __shared__ __bf16 sBt1[BN * LDBT];   // [n][k]
    __shared__ __bf16 sBt2[BN * LDBT];

    const int tid  = threadIdx.x;
    const int wave = tid >> 5;          // 0..7
    const int lane = tid & 31;
    const int wy   = wave >> 1;         // 0..3 -> M offset 32*wy
    const int wx   = wave & 1;          // 0..1 -> N offset 32*wx

    const int m0 = blockIdx.y * BM;
    const int n0 = blockIdx.x * BN;

    const int lrow  = lane & 15;
    const int khalf = (lane < 16) ? 0 : 8;   // ISA 16-bit lane K-half

    v8f acc[2][2] = {};   // [tm][tn]

    for (int k0 = 0; k0 < K; k0 += BK) {
        // ---- stage A tiles: BM x BK, float4 per thread x4 (global_load_b128) ----
        #pragma unroll
        for (int i = 0; i < (BM * BK) / (256 * 4); ++i) {   // 4 iters
            int idx4 = tid + i * 256;        // each covers 4 elements
            int r  = idx4 >> 3;              // BK/4 = 8 chunks per row
            int c4 = (idx4 & 7) * 4;
            int gm = m0 + r;
            float4 a1 = make_float4(0.f, 0.f, 0.f, 0.f), a2 = a1;
            if (gm < M) {
                a1 = *(const float4*)(A1 + (long)gm * K + k0 + c4);
                a2 = *(const float4*)(A2 + (long)gm * K + k0 + c4);
            }
            *(v4bf*)&sA1[r * LDA + c4] = cvt4(a1);
            *(v4bf*)&sA2[r * LDA + c4] = cvt4(a2);
        }
        // ---- stage B tiles transposed: read float4 along N, scatter 4 rows ----
        #pragma unroll
        for (int i = 0; i < (BK * BN) / (256 * 4); ++i) {   // 2 iters
            int idx4 = tid + i * 256;
            int r  = idx4 >> 4;              // BN/4 = 16 chunks per k-row
            int c4 = (idx4 & 15) * 4;
            float4 b1 = *(const float4*)(B1 + (long)(k0 + r) * N + n0 + c4);
            float4 b2 = *(const float4*)(B2 + (long)(k0 + r) * N + n0 + c4);
            sBt1[(c4 + 0) * LDBT + r] = (__bf16)b1.x;
            sBt1[(c4 + 1) * LDBT + r] = (__bf16)b1.y;
            sBt1[(c4 + 2) * LDBT + r] = (__bf16)b1.z;
            sBt1[(c4 + 3) * LDBT + r] = (__bf16)b1.w;
            sBt2[(c4 + 0) * LDBT + r] = (__bf16)b2.x;
            sBt2[(c4 + 1) * LDBT + r] = (__bf16)b2.y;
            sBt2[(c4 + 2) * LDBT + r] = (__bf16)b2.z;
            sBt2[(c4 + 3) * LDBT + r] = (__bf16)b2.w;
        }
        __syncthreads();

        // prefetch next K tile of A (global_prefetch_b8)
        if (k0 + BK < K) {
            int gm = m0 + (tid >> 1);
            if (gm < M) {
                __builtin_prefetch(&A1[(long)gm * K + k0 + BK], 0, 1);
                __builtin_prefetch(&A2[(long)gm * K + k0 + BK], 0, 1);
            }
        }

        // ---- build fragments: 2x ds_load_b128 each ----
        v16bf fa1[2], fa2[2], fb1[2], fb2[2];
        #pragma unroll
        for (int tm = 0; tm < 2; ++tm) {
            int arow = wy * 32 + tm * 16 + lrow;
            fa1[tm] = load_frag(&sA1[arow * LDA], khalf);
            fa2[tm] = load_frag(&sA2[arow * LDA], khalf);
        }
        #pragma unroll
        for (int tn = 0; tn < 2; ++tn) {
            int brow = wx * 32 + tn * 16 + lrow;   // column n of B == row of Bt
            fb1[tn] = load_frag(&sBt1[brow * LDBT], khalf);
            fb2[tn] = load_frag(&sBt2[brow * LDBT], khalf);
        }

        // ---- 8 WMMAs per K step ----
        #pragma unroll
        for (int tm = 0; tm < 2; ++tm)
            #pragma unroll
            for (int tn = 0; tn < 2; ++tn) {
                acc[tm][tn] = __builtin_amdgcn_wmma_f32_16x16x32_bf16(
                    false, fa1[tm], false, fb1[tn], (short)0, acc[tm][tn], false, false);
                acc[tm][tn] = __builtin_amdgcn_wmma_f32_16x16x32_bf16(
                    false, fa2[tm], false, fb2[tn], (short)0, acc[tm][tn], false, false);
            }
        __syncthreads();
    }

    // ---- store: C/D f32 layout: lane n = lane&15, VGPR r -> M = r + (lane<16?0:8)
    #pragma unroll
    for (int tm = 0; tm < 2; ++tm) {
        #pragma unroll
        for (int tn = 0; tn < 2; ++tn) {
            int coln = n0 + wx * 32 + tn * 16 + (lane & 15);
            float bv = bias[coln];
            #pragma unroll
            for (int r = 0; r < 8; ++r) {
                int rowm = m0 + wy * 32 + tm * 16 + r + ((lane < 16) ? 0 : 8);
                if (rowm < M) {
                    float v = acc[tm][tn][r] + bv;
                    if (RELU) v = fmaxf(v, 0.0f);
                    C[(long)rowm * N + coln] = v;
                }
            }
        }
    }
}

// =====================================================================
// graph mean-pool: sums[batch[n]] += h[n], counts[batch[n]] += 1
// =====================================================================
__global__ void k_pool_sum(const float* __restrict__ h, const int* __restrict__ batch,
                           float* __restrict__ sums, float* __restrict__ cnt) {
    long tid = (long)blockIdx.x * blockDim.x + threadIdx.x;
    long total = (long)N_NODES * (HID / 4);
    if (tid >= total) return;
    int node = (int)(tid >> 6);
    int c    = (int)(tid & 63);
    int g = batch[node];
    const float4 v = *(const float4*)(h + (long)node * HID + c * 4);
    float* p = sums + (long)g * HID + c * 4;
    atomicAdd(p + 0, v.x);
    atomicAdd(p + 1, v.y);
    atomicAdd(p + 2, v.z);
    atomicAdd(p + 3, v.w);
    if (c == 0) atomicAdd(&cnt[g], 1.0f);
}

__global__ void k_pool_div(float* __restrict__ sums, const float* __restrict__ cnt) {
    int i = blockIdx.x * blockDim.x + threadIdx.x;
    if (i >= N_GRAPHS * HID) return;
    int g = i / HID;
    sums[i] = sums[i] / fmaxf(cnt[g], 1.0f);
}

// =====================================================================
// tiny head: out[g,j] = pooled[g] . W[:,j] + b[j]   (128x10, scalar)
// =====================================================================
__global__ void k_final_linear(const float* __restrict__ pooled,
                               const float* __restrict__ W,
                               const float* __restrict__ b,
                               float* __restrict__ out) {
    int tid = blockIdx.x * blockDim.x + threadIdx.x;
    if (tid >= N_GRAPHS * N_CLASSES) return;
    int g = tid / N_CLASSES;
    int j = tid % N_CLASSES;
    float s = b[j];
    #pragma unroll 4
    for (int k = 0; k < HID; ++k)
        s += pooled[g * HID + k] * W[k * N_CLASSES + j];
    out[tid] = s;
}

// =====================================================================
// launch
// =====================================================================
extern "C" void kernel_launch(void* const* d_in, const int* in_sizes, int n_in,
                              void* d_out, int out_size, void* d_ws, size_t ws_size,
                              hipStream_t stream) {
    const float* x     = (const float*)d_in[0];
    const int*   ei    = (const int*)  d_in[1];
    const int*   batch = (const int*)  d_in[2];
    const float* W1r   = (const float*)d_in[3];
    const float* W1e   = (const float*)d_in[4];
    const float* b1    = (const float*)d_in[5];
    const float* W2r   = (const float*)d_in[6];
    const float* W2e   = (const float*)d_in[7];
    const float* b2    = (const float*)d_in[8];
    const float* W3r   = (const float*)d_in[9];
    const float* W3e   = (const float*)d_in[10];
    const float* b3    = (const float*)d_in[11];
    const float* Wlin  = (const float*)d_in[12];
    const float* blin  = (const float*)d_in[13];
    float* out = (float*)d_out;

    const int* src = ei;             // edge_index[0]
    const int* dst = ei + N_EDGES;   // edge_index[1]

    // workspace carve-up (~154 MB)
    float* AGG  = (float*)d_ws;                       // [N_NODES, HID]
    float* HA   = AGG  + (size_t)N_NODES * HID;       // [N_NODES, HID]
    float* HB   = HA   + (size_t)N_NODES * HID;       // [N_NODES, HID]
    float* SUMS = HB   + (size_t)N_NODES * HID;       // [N_GRAPHS, HID]
    float* CNT  = SUMS + (size_t)N_GRAPHS * HID;      // [N_GRAPHS]

    const int T = 256;
    dim3 blk(T);
    dim3 gemm_grid(HID / BN, (N_NODES + BM - 1) / BM);

    // ---------------- layer 1 (K = D_IN = 128) ----------------
    {
        long n = (long)N_NODES * D_IN;
        k_zero<<<dim3((unsigned)((n + T - 1) / T)), blk, 0, stream>>>(AGG, n);
        long work = (long)N_EDGES * (D_IN / 4);
        k_scatter_add<5><<<dim3((unsigned)((work + T - 1) / T)), blk, 0, stream>>>(
            x, src, dst, AGG, N_EDGES, D_IN);
        k_dual_gemm_bias<true><<<gemm_grid, blk, 0, stream>>>(
            x, AGG, W1r, W1e, b1, HA, N_NODES, D_IN, HID);
    }
    // ---------------- layer 2 (K = HID = 256) ----------------
    {
        long n = (long)N_NODES * HID;
        k_zero<<<dim3((unsigned)((n + T - 1) / T)), blk, 0, stream>>>(AGG, n);
        long work = (long)N_EDGES * (HID / 4);
        k_scatter_add<6><<<dim3((unsigned)((work + T - 1) / T)), blk, 0, stream>>>(
            HA, src, dst, AGG, N_EDGES, HID);
        k_dual_gemm_bias<true><<<gemm_grid, blk, 0, stream>>>(
            HA, AGG, W2r, W2e, b2, HB, N_NODES, HID, HID);
    }
    // ---------------- layer 3 (no relu) ----------------
    {
        long n = (long)N_NODES * HID;
        k_zero<<<dim3((unsigned)((n + T - 1) / T)), blk, 0, stream>>>(AGG, n);
        long work = (long)N_EDGES * (HID / 4);
        k_scatter_add<6><<<dim3((unsigned)((work + T - 1) / T)), blk, 0, stream>>>(
            HB, src, dst, AGG, N_EDGES, HID);
        k_dual_gemm_bias<false><<<gemm_grid, blk, 0, stream>>>(
            HB, AGG, W3r, W3e, b3, HA, N_NODES, HID, HID);
    }
    // ---------------- mean pool + head ----------------
    {
        long n = (long)N_GRAPHS * HID + N_GRAPHS;   // SUMS + CNT contiguous
        k_zero<<<dim3((unsigned)((n + T - 1) / T)), blk, 0, stream>>>(SUMS, n);
        long work = (long)N_NODES * (HID / 4);
        k_pool_sum<<<dim3((unsigned)((work + T - 1) / T)), blk, 0, stream>>>(
            HA, batch, SUMS, CNT);
        k_pool_div<<<dim3((N_GRAPHS * HID + T - 1) / T), blk, 0, stream>>>(SUMS, CNT);
        k_final_linear<<<dim3((N_GRAPHS * N_CLASSES + T - 1) / T), blk, 0, stream>>>(
            SUMS, Wlin, blin, out);
    }
}
